// SimCSE_52132313038850
// MI455X (gfx1250) — matrix operand order repaired and checked by reading, not compiled, not run on hardware
//
#include <hip/hip_runtime.h>
#include <hip/hip_bf16.h>
#include <math.h>

typedef __attribute__((ext_vector_type(16))) _Float16 v16h;
typedef __attribute__((ext_vector_type(8)))  _Float16 v8h;
typedef __attribute__((ext_vector_type(8)))  float    v8f;

#define N_ROWS 8192
#define DIM    512
#define INV_T  20.0f   // 1 / 0.05
#define EPS_N  1e-8f
#define WAVES_PER_BLOCK 8

union AV { v16h v; v8h h[2]; };

// ---------------- Phase 1: row L2-normalize, fp32 -> f16 ----------------
// One wave (32 lanes) per row; lane covers 16 contiguous elements.
__global__ void __launch_bounds__(256) normalize_f16(
    const float* __restrict__ q, const float* __restrict__ p,
    _Float16* __restrict__ qn, _Float16* __restrict__ pn) {
  const int gwave = (blockIdx.x * blockDim.x + threadIdx.x) >> 5;  // 0..16383
  const int lane  = threadIdx.x & 31;

  const float* src;
  _Float16*    dst;
  if (gwave < N_ROWS) {
    src = q  + (size_t)gwave * DIM;
    dst = qn + (size_t)gwave * DIM;
  } else {
    src = p  + (size_t)(gwave - N_ROWS) * DIM;
    dst = pn + (size_t)(gwave - N_ROWS) * DIM;
  }

  float vals[16];
  const float4* s4 = (const float4*)(src + lane * 16);
  float ss = 0.0f;
#pragma unroll
  for (int i = 0; i < 4; ++i) {
    float4 t = s4[i];
    vals[i*4+0] = t.x; vals[i*4+1] = t.y; vals[i*4+2] = t.z; vals[i*4+3] = t.w;
    ss += t.x*t.x + t.y*t.y + t.z*t.z + t.w*t.w;
  }
#pragma unroll
  for (int off = 16; off >= 1; off >>= 1)
    ss += __shfl_xor(ss, off, 32);

  const float scale = 1.0f / fmaxf(sqrtf(ss), EPS_N);

  v8h h0, h1;
#pragma unroll
  for (int i = 0; i < 8; ++i) {
    h0[i] = (_Float16)(vals[i]     * scale);
    h1[i] = (_Float16)(vals[i + 8] * scale);
  }
  *(v8h*)(dst + lane * 16)     = h0;
  *(v8h*)(dst + lane * 16 + 8) = h1;
}

// ---------------- Phase 2: WMMA GEMM with fused online logsumexp ----------------
// One block per 16-row tile of queries (512 blocks). The 8 waves of a block
// split the 8192-column sweep (2 x 16-col tiles per iteration each), keep
// per-lane running (max, sumexp) + the diagonal, then merge partials via LDS.
__global__ void __launch_bounds__(256) simcse_lse(
    const _Float16* __restrict__ qn, const _Float16* __restrict__ pn,
    float* __restrict__ rowloss) {
  const int lane  = threadIdx.x & 31;
  const int wid   = threadIdx.x >> 5;             // wave within block, 0..7
  const int rbase = blockIdx.x * 16;              // 16 query rows per block
  const int halfL = lane & 15;
  const int hi    = lane >> 4;                    // 0: rows rbase+0..7, 1: +8..15

  __shared__ float lds_m[WAVES_PER_BLOCK][16];
  __shared__ float lds_s[WAVES_PER_BLOCK][16];
  __shared__ float lds_d[WAVES_PER_BLOCK][16];

  // A: row = rbase + (lane&15); lanes 0-15 hold K 0-15, lanes 16-31 K 16-31
  const _Float16* aBase = qn + (size_t)(rbase + halfL) * DIM + (hi ? 16 : 0);
  // B: col n = lane&15 reads pn row (cb + n); lanes 0-15 K 0-15, lanes 16-31 K 16-31
  const size_t bLaneOff = (size_t)(hi ? 16 : 0);

  float m[8], s[8], diag[8];
#pragma unroll
  for (int v = 0; v < 8; ++v) { m[v] = -INFINITY; s[v] = 0.0f; diag[v] = 0.0f; }

  // wave `wid` handles columns [wid*32 + j*256, +32) for j = 0..31
  for (int cb = wid * 32; cb < N_ROWS; cb += WAVES_PER_BLOCK * 32) {
    const _Float16* b0Base = pn + (size_t)(cb + halfL)      * DIM + bLaneOff;
    const _Float16* b1Base = pn + (size_t)(cb + 16 + halfL) * DIM + bLaneOff;

    v8f c0 = {}; v8f c1 = {};
#pragma unroll 8
    for (int k = 0; k < DIM; k += 32) {
      AV a;
      a.v = *(const v16h*)(aBase + k);
      v16h b0 = *(const v16h*)(b0Base + k);
      v16h b1 = *(const v16h*)(b1Base + k);
      c0 = __builtin_amdgcn_wmma_f32_16x16x32_f16(false, a.v, false, b0,
                                                  (short)0, c0, false, false);
      c1 = __builtin_amdgcn_wmma_f32_16x16x32_f16(false, a.v, false, b1,
                                                  (short)0, c1, false, false);
    }

    // online logsumexp update; C layout: c[v] = sim[rbase + v + 8*hi][col]
#pragma unroll
    for (int v = 0; v < 8; ++v) {
      const int rowg = rbase + v + (hi << 3);
      float x0 = c0[v] * INV_T;
      float x1 = c1[v] * INV_T;
      if (cb + halfL      == rowg) diag[v] = x0;
      if (cb + 16 + halfL == rowg) diag[v] = x1;
      float mn = fmaxf(m[v], x0);
      s[v] = s[v] * __expf(m[v] - mn) + __expf(x0 - mn);
      m[v] = mn;
      mn = fmaxf(m[v], x1);
      s[v] = s[v] * __expf(m[v] - mn) + __expf(x1 - mn);
      m[v] = mn;
    }
  }

  // merge the 16 column-partials within each half-wave (xor 1,2,4,8 stays in half)
#pragma unroll
  for (int v = 0; v < 8; ++v) {
#pragma unroll
    for (int off = 8; off >= 1; off >>= 1) {
      float mo = __shfl_xor(m[v], off, 32);
      float so = __shfl_xor(s[v], off, 32);
      float mn = fmaxf(m[v], mo);
      s[v] = s[v] * __expf(m[v] - mn) + so * __expf(mo - mn);
      m[v] = mn;
      diag[v] += __shfl_xor(diag[v], off, 32);
    }
  }

  // publish this wave's 16-row partials
  if (lane == 0) {
#pragma unroll
    for (int v = 0; v < 8; ++v) {
      lds_m[wid][v] = m[v]; lds_s[wid][v] = s[v]; lds_d[wid][v] = diag[v];
    }
  } else if (lane == 16) {
#pragma unroll
    for (int v = 0; v < 8; ++v) {
      lds_m[wid][v + 8] = m[v]; lds_s[wid][v + 8] = s[v]; lds_d[wid][v + 8] = diag[v];
    }
  }
  __syncthreads();

  // cross-wave logsumexp merge: one thread per row
  if (threadIdx.x < 16) {
    const int r = threadIdx.x;
    float M = lds_m[0][r], S = lds_s[0][r], Dg = lds_d[0][r];
#pragma unroll
    for (int w = 1; w < WAVES_PER_BLOCK; ++w) {
      float mo = lds_m[w][r], so = lds_s[w][r];
      float mn = fmaxf(M, mo);
      S = S * __expf(M - mn) + so * __expf(mo - mn);
      M = mn;
      Dg += lds_d[w][r];
    }
    rowloss[rbase + r] = M + logf(S) - Dg;
  }
}

// ---------------- Phase 3: mean over rows ----------------
__global__ void __launch_bounds__(256) reduce_mean(
    const float* __restrict__ rowloss, float* __restrict__ out) {
  __shared__ float sm[256];
  float s = 0.0f;
  for (int i = threadIdx.x; i < N_ROWS; i += 256) s += rowloss[i];
  sm[threadIdx.x] = s;
  __syncthreads();
  for (int stride = 128; stride >= 1; stride >>= 1) {
    if ((int)threadIdx.x < stride) sm[threadIdx.x] += sm[threadIdx.x + stride];
    __syncthreads();
  }
  if (threadIdx.x == 0) out[0] = sm[0] * (1.0f / (float)N_ROWS);
}

extern "C" void kernel_launch(void* const* d_in, const int* in_sizes, int n_in,
                              void* d_out, int out_size, void* d_ws, size_t ws_size,
                              hipStream_t stream) {
  const float* q = (const float*)d_in[0];
  const float* p = (const float*)d_in[1];
  float* out = (float*)d_out;

  _Float16* qn = (_Float16*)d_ws;
  _Float16* pn = qn + (size_t)N_ROWS * DIM;
  float* rowloss = (float*)(pn + (size_t)N_ROWS * DIM);

  // Phase 1: 16384 rows, 1 wave/row, 8 waves/block -> 2048 blocks
  normalize_f16<<<2048, 256, 0, stream>>>(q, p, qn, pn);

  // Phase 2: 512 blocks (one 16-row tile each), 8 column-split waves/block
  simcse_lse<<<512, 256, 0, stream>>>(qn, pn, rowloss);

  // Phase 3: single-block mean
  reduce_mean<<<1, 256, 0, stream>>>(rowloss, out);
}
